// Decoder_83150566850653
// MI455X (gfx1250) — matrix operand order repaired and checked by reading, not compiled, not run on hardware
//
#include <hip/hip_runtime.h>
#include <hip/hip_bf16.h>

// ---------------------------------------------------------------------------
// CDNA5 (gfx1250) attention-LSTM decoder.
//  - All GEMMs (K-proj, gates, out_o, vocab logits) use v_wmma_f32_16x16x32_f16.
//  - Weights staged once per launch into f16 [N][K] (transposed) so both A and
//    B fragments are contiguous 16B/32B global_load_b128s.
//  - Serial per-step chain handled with stream-ordered kernels (graph-safe).
// ---------------------------------------------------------------------------

typedef _Float16 half_t;
typedef __attribute__((ext_vector_type(16))) _Float16 v16h;
typedef __attribute__((ext_vector_type(8)))  _Float16 v8h;
typedef __attribute__((ext_vector_type(8)))  float    v8f;

#define DEV static __device__ __forceinline__

namespace cfg {
constexpr int B    = 64;
constexpr int P    = 144;
constexpr int DENC = 1792;
constexpr int DDEC = 512;
constexpr int DATT = 512;
constexpr int E    = 300;
constexpr int V    = 32000;
constexpr int T    = 30;
constexpr int KGR  = 2604;   // 512 + 300 + 1792
constexpr int KG   = 2624;   // padded to multiple of 32
constexpr int ZOFF = 812;    // 512 + 300
constexpr int KO   = 320;    // 300 padded to multiple of 32 (out_o width)
constexpr int KHZ  = 2304;   // 512 + 1792 (h_new | z), multiple of 32
}

// ---------------- WMMA fragment loaders (ISA 7.12.2 layouts) ----------------
// A (16x32 f16, MxK): lane L holds row M=L%16.
//   elements 0..7  -> K = k0 + (L/16)*8 + e
//   elements 8..15 -> K = k0 + 16 + (L/16)*8 + (e-8)
DEV v16h load_a_frag(const half_t* A, int ld, int m0, int k0, int lane) {
  int r = m0 + (lane & 15);
  int h = (lane >> 4) << 3;                       // 0 or 8
  const half_t* p = A + (size_t)r * ld + k0 + h;
  v8h lo = *(const v8h*)(p);
  v8h hi = *(const v8h*)(p + 16);
  return __builtin_shufflevector(lo, hi, 0,1,2,3,4,5,6,7,8,9,10,11,12,13,14,15);
}
// B (32x16 f16, KxN) from a transposed [N][K] buffer: lane L holds col N=L%16,
//   elements 0..7  -> K = k0 + (L/16)*8 + e
//   elements 8..15 -> K = k0 + 16 + (L/16)*8 + (e-8)
DEV v16h load_bT_frag(const half_t* Bt, int ld, int n0, int k0, int lane) {
  int c = n0 + (lane & 15);
  int h = (lane >> 4) << 3;                       // 0 or 8
  const half_t* p = Bt + (size_t)c * ld + k0 + h;
  v8h lo = *(const v8h*)(p);
  v8h hi = *(const v8h*)(p + 16);
  return __builtin_shufflevector(lo, hi, 0,1,2,3,4,5,6,7,8,9,10,11,12,13,14,15);
}
DEV v8f wmma_f16(v16h a, v16h b, v8f c) {
  return __builtin_amdgcn_wmma_f32_16x16x32_f16(false, a, false, b, (short)0, c,
                                                false, false);
}
// D: element r (0..7) of lane L -> M = m0 + r + 8*(L/16), N = n0 + (L%16)

DEV float sigm(float x) { return 1.f / (1.f + expf(-x)); }

// blockDim.x == 256 reductions
DEV float block_sum(float v, float* red) {
  int tid = threadIdx.x;
  red[tid] = v; __syncthreads();
  for (int s = 128; s > 0; s >>= 1) { if (tid < s) red[tid] += red[tid + s]; __syncthreads(); }
  float r = red[0]; __syncthreads();
  return r;
}

// ---------------------------------------------------------------------------
// One-time kernels
// ---------------------------------------------------------------------------
__global__ __launch_bounds__(256) void k_zero_t0(float* pred, float* alph) {
  using namespace cfg;
  long i = (long)blockIdx.x * 256 + threadIdx.x;
  if (i < (long)B * V) { int b = (int)(i / V); int v = (int)(i % V); pred[(long)b * T * V + v] = 0.f; }
  if (i < (long)B * P) { int b = (int)(i / P); int p = (int)(i % P); alph[(long)b * T * P + p] = 0.f; }
}

__global__ __launch_bounds__(256) void k_wkT(const float* Wk, half_t* WkT) {
  using namespace cfg;
  long i = (long)blockIdx.x * 256 + threadIdx.x;       // over DATT*DENC
  if (i >= (long)DATT * DENC) return;
  int a = (int)(i / DENC), d = (int)(i % DENC);
  WkT[i] = (half_t)Wk[(size_t)d * DATT + a];
}

__global__ __launch_bounds__(256) void k_wgT(const float* Wh, const float* Wu,
                                             const float* Wa, half_t* WgT) {
  using namespace cfg;
  long i = (long)blockIdx.x * 256 + threadIdx.x;       // over 2048*KG
  if (i >= (long)4 * DDEC * KG) return;
  int n = (int)(i / KG), k = (int)(i % KG);
  float v = 0.f;
  if      (k < DDEC)  v = Wh[(size_t)k * (4 * DDEC) + n];
  else if (k < ZOFF)  v = Wu[(size_t)(k - DDEC) * (4 * DDEC) + n];
  else if (k < KGR)   v = Wa[(size_t)(k - ZOFF) * (4 * DDEC) + n];
  WgT[i] = (half_t)v;
}

__global__ __launch_bounds__(256) void k_wloT(const float* W_Lo, half_t* WLoT) {
  using namespace cfg;
  long i = (long)blockIdx.x * 256 + threadIdx.x;       // over V*KO
  if (i >= (long)V * KO) return;
  int n = (int)(i / KO), k = (int)(i % KO);
  WLoT[i] = (half_t)((k < E) ? W_Lo[(size_t)k * V + n] : 0.f);
}

// [W_Lh ; W_Lz] -> transposed f16 [KO][KHZ]; rows (out dims) >= E are zero so
// the padded out_o columns used by the logits GEMM are exactly zero.
__global__ __launch_bounds__(256) void k_wlhzT(const float* W_Lh, const float* W_Lz,
                                               half_t* WLhzT) {
  using namespace cfg;
  long i = (long)blockIdx.x * 256 + threadIdx.x;       // over KO*KHZ
  if (i >= (long)KO * KHZ) return;
  int n = (int)(i / KHZ), k = (int)(i % KHZ);
  float v = 0.f;
  if (n < E) v = (k < DDEC) ? W_Lh[(size_t)k * E + n] : W_Lz[(size_t)(k - DDEC) * E + n];
  WLhzT[i] = (half_t)v;
}

// LN over encoder dim; one block per (b,p) row; output f16.
__global__ __launch_bounds__(256) void k_ln_feats(const float* feats, const float* g,
                                                  const float* bia, half_t* out) {
  using namespace cfg;
  __shared__ float red[256];
  const float* x = feats + (size_t)blockIdx.x * DENC;
  float s = 0.f;
  for (int d = threadIdx.x; d < DENC; d += 256) s += x[d];
  float mean = block_sum(s, red) * (1.f / DENC);
  float v = 0.f;
  for (int d = threadIdx.x; d < DENC; d += 256) { float t = x[d] - mean; v += t * t; }
  float var = block_sum(v, red) * (1.f / DENC);
  float rstd = rsqrtf(var + 1e-5f);
  half_t* o = out + (size_t)blockIdx.x * DENC;
  for (int d = threadIdx.x; d < DENC; d += 256)
    o[d] = (half_t)((x[d] - mean) * rstd * g[d] + bia[d]);
}

// mean over P, then h0/c0 projections; one block per batch row.
__global__ __launch_bounds__(256) void k_init_state(const float* feats,
                                                    const float* W_ih, const float* b_ih,
                                                    const float* W_ic, const float* b_ic,
                                                    float* h, float* c) {
  using namespace cfg;
  __shared__ float mean[DENC];
  int b = blockIdx.x;
  for (int d = threadIdx.x; d < DENC; d += 256) {
    float s = 0.f;
    const float* p = feats + (size_t)b * P * DENC + d;
    for (int pp = 0; pp < P; ++pp) s += p[(size_t)pp * DENC];
    mean[d] = s * (1.f / P);
  }
  __syncthreads();
  for (int j = threadIdx.x; j < DDEC; j += 256) {
    float sh = 0.f, sc = 0.f;
    for (int d = 0; d < DENC; ++d) {
      float m = mean[d];
      sh += m * W_ih[(size_t)d * DDEC + j];
      sc += m * W_ic[(size_t)d * DDEC + j];
    }
    h[b * DDEC + j] = sh + b_ih[j];
    c[b * DDEC + j] = sc + b_ic[j];
  }
}

// K = feats_n @ Wk + bk : M=9216, N=512, K=1792. grid=(4,576), 8 waves/block.
__global__ __launch_bounds__(256) void k_proj_K(const half_t* featsh, const half_t* WkT,
                                                const float* bk, float* Kmat) {
  using namespace cfg;
  int lane = threadIdx.x & 31;
  int wave = threadIdx.x >> 5;
  int n0 = (blockIdx.x * 8 + wave) * 16;
  int m0 = blockIdx.y * 16;
  v8f acc = {};
  for (int k0 = 0; k0 < DENC; k0 += 32) {
    v16h a = load_a_frag(featsh, DENC, m0, k0, lane);
    v16h b = load_bT_frag(WkT, DENC, n0, k0, lane);
    acc = wmma_f16(a, b, acc);
  }
  int col = lane & 15, hm = (lane >> 4) * 8;
#pragma unroll
  for (int r = 0; r < 8; ++r) {
    int m = m0 + hm + r, n = n0 + col;
    Kmat[(size_t)m * DATT + n] = acc[r] + bk[n];
  }
}

// ---------------------------------------------------------------------------
// Per-step kernels
// ---------------------------------------------------------------------------
// Attention: one block per batch row. Produces alpha (to d_out) and the f16
// A-rows: xcat = [h | emb_t | z | 0pad] (gates GEMM) and hz16 z-part (out_o GEMM).
__global__ __launch_bounds__(256) void k_attn(const float* h, const float* lnqg,
                                              const float* lnqb, const float* Wq,
                                              const float* bq, const float* Wbeta,
                                              const float* bbeta, const float* Kmat,
                                              const half_t* featsh, const int* captions,
                                              const float* emb_table, half_t* hz16,
                                              half_t* xcat, float* alph, int t) {
  using namespace cfg;
  __shared__ float hq[DDEC];
  __shared__ float Qs[DATT];
  __shared__ float sc[P];
  __shared__ float red[256];
  __shared__ float sbeta;
  int b = blockIdx.x, tid = threadIdx.x;

  // LayerNorm(h)
  float s = 0.f;
  for (int j = tid; j < DDEC; j += 256) s += h[b * DDEC + j];
  float mean = block_sum(s, red) * (1.f / DDEC);
  float v = 0.f;
  for (int j = tid; j < DDEC; j += 256) { float d = h[b * DDEC + j] - mean; v += d * d; }
  float rstd = rsqrtf(block_sum(v, red) * (1.f / DDEC) + 1e-5f);
  for (int j = tid; j < DDEC; j += 256)
    hq[j] = (h[b * DDEC + j] - mean) * rstd * lnqg[j] + lnqb[j];
  __syncthreads();

  // Q = hq @ Wq + bq
  for (int a = tid; a < DATT; a += 256) {
    float acc = bq[a];
    for (int k = 0; k < DDEC; ++k) acc += hq[k] * Wq[(size_t)k * DATT + a];
    Qs[a] = acc;
  }
  // beta = sigmoid(hq @ Wbeta + bbeta)
  float pb = 0.f;
  for (int k = tid; k < DDEC; k += 256) pb += hq[k] * Wbeta[k];
  pb = block_sum(pb, red);
  if (tid == 0) sbeta = sigm(pb + bbeta[0]);
  __syncthreads();

  // scores + softmax over P
  const float scale = rsqrtf((float)DATT);
  if (tid < P) {
    const float* kr = Kmat + ((size_t)b * P + tid) * DATT;
    float acc = 0.f;
    for (int a = 0; a < DATT; ++a) acc += Qs[a] * kr[a];
    sc[tid] = acc * scale;
  }
  __syncthreads();
  red[tid] = (tid < P) ? sc[tid] : -1e30f;
  __syncthreads();
  for (int st = 128; st > 0; st >>= 1) { if (tid < st) red[tid] = fmaxf(red[tid], red[tid + st]); __syncthreads(); }
  float mx = red[0]; __syncthreads();
  float e = (tid < P) ? expf(sc[tid] - mx) : 0.f;
  float se = block_sum(e, red);
  if (tid < P) {
    float a = e / se;
    sc[tid] = a;
    alph[((size_t)b * T + t) * P + tid] = a;
  }
  __syncthreads();

  // context, z, and A-row assembly
  float beta = sbeta;
  const half_t* fb = featsh + (size_t)b * P * DENC;
  for (int d = tid; d < DENC; d += 256) {
    float acc = 0.f;
    for (int p = 0; p < P; ++p) acc += sc[p] * (float)fb[(size_t)p * DENC + d];
    half_t zz = (half_t)(beta * acc);
    xcat[(size_t)b * KG + ZOFF + d] = zz;     // gates GEMM A (z segment)
    hz16[(size_t)b * KHZ + DDEC + d] = zz;    // out_o GEMM A (z segment)
  }
  for (int j = tid; j < DDEC; j += 256) xcat[(size_t)b * KG + j] = (half_t)h[b * DDEC + j];
  int tok = captions[b * T + (t - 1)];
  for (int e2 = tid; e2 < E; e2 += 256)
    xcat[(size_t)b * KG + DDEC + e2] = (half_t)emb_table[(size_t)tok * E + e2];
  for (int k = KGR + tid; k < KG; k += 256) xcat[(size_t)b * KG + k] = (half_t)0.f;
}

// gates = xcat @ Wcat + b, then LSTM update. Each wave owns the i/f/o/g tiles
// of the same 16x16 (batch, j) patch -> update stays in registers.
// grid=(4,4), 8 waves/block: q = bx*8+wave in [0,32), m0 = by*16.
// Also writes h_new in f16 into hz16[:, 0:512] for the out_o WMMA GEMM.
__global__ __launch_bounds__(256) void k_gates_lstm(const half_t* xcat, const half_t* WgT,
                                                    const float* b_gates, float* h, float* c,
                                                    half_t* hz16) {
  using namespace cfg;
  int lane = threadIdx.x & 31;
  int wave = threadIdx.x >> 5;
  int q  = blockIdx.x * 8 + wave;     // j-tile within 512
  int m0 = blockIdx.y * 16;
  v8f acc[4] = {{}, {}, {}, {}};
  for (int k0 = 0; k0 < KG; k0 += 32) {
    v16h a = load_a_frag(xcat, KG, m0, k0, lane);
#pragma unroll
    for (int g = 0; g < 4; ++g) {
      v16h bf = load_bT_frag(WgT, KG, q * 16 + DDEC * g, k0, lane);
      acc[g] = wmma_f16(a, bf, acc[g]);
    }
  }
  int col = lane & 15, hm = (lane >> 4) * 8;
#pragma unroll
  for (int r = 0; r < 8; ++r) {
    int b = m0 + hm + r;
    int j = q * 16 + col;
    float i_t = acc[0][r] + b_gates[j];
    float f_t = acc[1][r] + b_gates[DDEC + j];
    float o_t = acc[2][r] + b_gates[2 * DDEC + j];
    float g_t = acc[3][r] + b_gates[3 * DDEC + j];
    float cv = c[b * DDEC + j];
    float cn = sigm(f_t) * cv + sigm(i_t) * tanhf(g_t);
    float hn = sigm(o_t) * tanhf(cn);
    c[b * DDEC + j] = cn;
    h[b * DDEC + j] = hn;
    hz16[(size_t)b * KHZ + j] = (half_t)hn;
  }
}

// out_o = emb + [h_new|z] @ [W_Lh;W_Lz] + biases, via WMMA.
// M=64, N=320 (300 + zero pad), K=2304. grid=10, 8 waves/block; wave w of
// block bx handles tile id = bx*8+w: nt = id/4 (n-tile), mt = id%4 (m-tile).
__global__ __launch_bounds__(256) void k_outo_wmma(const half_t* hz16, const half_t* WLhzT,
                                                   const int* captions, const float* emb_table,
                                                   const float* b_Lh, const float* b_Lz,
                                                   half_t* outo, int t) {
  using namespace cfg;
  int lane = threadIdx.x & 31;
  int wave = threadIdx.x >> 5;
  int id = blockIdx.x * 8 + wave;     // 0..79
  int nt = id >> 2, mt = id & 3;
  int n0 = nt * 16, m0 = mt * 16;
  v8f acc = {};
  for (int k0 = 0; k0 < KHZ; k0 += 32) {
    v16h a = load_a_frag(hz16, KHZ, m0, k0, lane);
    v16h bf = load_bT_frag(WLhzT, KHZ, n0, k0, lane);
    acc = wmma_f16(a, bf, acc);
  }
  int col = lane & 15, hm = (lane >> 4) * 8;
  int e = n0 + col;
  float extra = (e < E) ? (b_Lh[e] + b_Lz[e]) : 0.f;
#pragma unroll
  for (int r = 0; r < 8; ++r) {
    int b = m0 + hm + r;
    float val = 0.f;
    if (e < E) {
      int tok = captions[b * T + (t - 1)];
      val = acc[r] + extra + emb_table[(size_t)tok * E + e];
    }
    outo[b * KO + e] = (half_t)val;   // padded columns stay exactly zero
  }
}

// logits = outo @ W_Lo + b_Lo written straight into predictions[:, t, :].
// M=64, N=32000, K=320. grid=250, 8 waves/block; each wave: 1 n-tile x 4 m-tiles.
__global__ __launch_bounds__(256) void k_logits(const half_t* outo, const half_t* WLoT,
                                                const float* b_Lo, float* pred, int t) {
  using namespace cfg;
  int lane = threadIdx.x & 31;
  int wave = threadIdx.x >> 5;
  int n0 = (blockIdx.x * 8 + wave) * 16;
  v8f acc[4] = {{}, {}, {}, {}};
  for (int k0 = 0; k0 < KO; k0 += 32) {
    v16h bf = load_bT_frag(WLoT, KO, n0, k0, lane);
#pragma unroll
    for (int mt = 0; mt < 4; ++mt) {
      v16h a = load_a_frag(outo, KO, mt * 16, k0, lane);
      acc[mt] = wmma_f16(a, bf, acc[mt]);
    }
  }
  int col = lane & 15, hm = (lane >> 4) * 8;
  int vv = n0 + col;
  float bias = b_Lo[vv];
#pragma unroll
  for (int mt = 0; mt < 4; ++mt) {
#pragma unroll
    for (int r = 0; r < 8; ++r) {
      int b = mt * 16 + hm + r;
      pred[((size_t)b * T + t) * V + vv] = acc[mt][r] + bias;
    }
  }
}

// ---------------------------------------------------------------------------
// Host-side launcher
// ---------------------------------------------------------------------------
static inline size_t alignup(size_t x) { return (x + 255) & ~(size_t)255; }

extern "C" void kernel_launch(void* const* d_in, const int* in_sizes, int n_in,
                              void* d_out, int out_size, void* d_ws, size_t ws_size,
                              hipStream_t stream) {
  using namespace cfg;
  const float* features = (const float*)d_in[0];
  const int*   captions = (const int*)d_in[1];
  const float* emb      = (const float*)d_in[2];
  const float* ln_q_g   = (const float*)d_in[3];
  const float* ln_q_b   = (const float*)d_in[4];
  const float* ln_kv_g  = (const float*)d_in[5];
  const float* ln_kv_b  = (const float*)d_in[6];
  const float* Wq   = (const float*)d_in[7];   const float* bq    = (const float*)d_in[8];
  const float* Wk   = (const float*)d_in[9];   const float* bk    = (const float*)d_in[10];
  const float* Wbt  = (const float*)d_in[11];  const float* bbt   = (const float*)d_in[12];
  const float* Wh   = (const float*)d_in[13];  const float* Wu    = (const float*)d_in[14];
  const float* Wa   = (const float*)d_in[15];  const float* b_g   = (const float*)d_in[16];
  const float* W_Lh = (const float*)d_in[17];  const float* b_Lh  = (const float*)d_in[18];
  const float* W_Lz = (const float*)d_in[19];  const float* b_Lz  = (const float*)d_in[20];
  const float* W_Lo = (const float*)d_in[21];  const float* b_Lo  = (const float*)d_in[22];
  const float* W_ih = (const float*)d_in[23];  const float* b_ih  = (const float*)d_in[24];
  const float* W_ic = (const float*)d_in[25];  const float* b_ic  = (const float*)d_in[26];

  float* pred = (float*)d_out;                       // [B,T,V] f32
  float* alph = pred + (size_t)B * T * V;            // [B,T,P] f32

  // workspace carve-up (~88 MB total)
  char* w = (char*)d_ws;
  size_t off = 0;
  half_t* featsh = (half_t*)(w + off); off = alignup(off + (size_t)B * P * DENC * 2);
  float*  Kmat   = (float*) (w + off); off = alignup(off + (size_t)B * P * DATT * 4);
  half_t* WkT    = (half_t*)(w + off); off = alignup(off + (size_t)DATT * DENC * 2);
  half_t* WgT    = (half_t*)(w + off); off = alignup(off + (size_t)4 * DDEC * KG * 2);
  half_t* WLoT   = (half_t*)(w + off); off = alignup(off + (size_t)V * KO * 2);
  half_t* WLhzT  = (half_t*)(w + off); off = alignup(off + (size_t)KO * KHZ * 2);
  float*  hbuf   = (float*) (w + off); off = alignup(off + (size_t)B * DDEC * 4);
  float*  cbuf   = (float*) (w + off); off = alignup(off + (size_t)B * DDEC * 4);
  half_t* hz16   = (half_t*)(w + off); off = alignup(off + (size_t)B * KHZ * 2);
  half_t* xcat   = (half_t*)(w + off); off = alignup(off + (size_t)B * KG * 2);
  half_t* outo   = (half_t*)(w + off); off = alignup(off + (size_t)B * KO * 2);
  (void)ws_size; (void)in_sizes; (void)n_in; (void)out_size;

  // t=0 output slices must be zero (d_out is poisoned by the harness)
  {
    long n = (long)B * V;
    k_zero_t0<<<dim3((unsigned)((n + 255) / 256)), 256, 0, stream>>>(pred, alph);
  }
  // one-time weight staging (f16, transposed [N][K])
  k_wkT  <<<dim3((DATT * DENC + 255) / 256), 256, 0, stream>>>(Wk, WkT);
  k_wgT  <<<dim3((4 * DDEC * KG + 255) / 256), 256, 0, stream>>>(Wh, Wu, Wa, WgT);
  k_wloT <<<dim3((V * KO + 255) / 256), 256, 0, stream>>>(W_Lo, WLoT);
  k_wlhzT<<<dim3((KO * KHZ + 255) / 256), 256, 0, stream>>>(W_Lh, W_Lz, WLhzT);
  // LN(features) -> f16, init h0/c0, K projection (WMMA)
  k_ln_feats<<<dim3(B * P), 256, 0, stream>>>(features, ln_kv_g, ln_kv_b, featsh);
  k_init_state<<<dim3(B), 256, 0, stream>>>(features, W_ih, b_ih, W_ic, b_ic, hbuf, cbuf);
  k_proj_K<<<dim3(DATT / (8 * 16), (B * P) / 16), 256, 0, stream>>>(featsh, WkT, bk, Kmat);

  // serial decode: 29 steps
  for (int t = 1; t < T; ++t) {
    k_attn<<<dim3(B), 256, 0, stream>>>(hbuf, ln_q_g, ln_q_b, Wq, bq, Wbt, bbt,
                                        Kmat, featsh, captions, emb, hz16, xcat, alph, t);
    k_gates_lstm<<<dim3(4, 4), 256, 0, stream>>>(xcat, WgT, b_g, hbuf, cbuf, hz16);
    k_outo_wmma<<<dim3(10), 256, 0, stream>>>(hz16, WLhzT, captions, emb,
                                              b_Lh, b_Lz, outo, t);
    k_logits<<<dim3(V / (8 * 16)), 256, 0, stream>>>(outo, WLoT, b_Lo, pred, t);
  }
}
